// BiasAttention_57475252355510
// MI455X (gfx1250) — matrix operand order repaired and checked by reading, not compiled, not run on hardware
//
#include <hip/hip_runtime.h>

#define H      8
#define Dh     32
#define NQ     1024
#define NK     1024
#define BIASD  128
#define KSPLIT 4
#define QSCALE 0.17677669529663687f  /* 32^-0.5 */

typedef __attribute__((ext_vector_type(16))) __bf16    v16bf;
typedef __attribute__((ext_vector_type(2)))  __bf16    v2bf;
typedef __attribute__((ext_vector_type(8)))  float     v8f;
typedef __attribute__((ext_vector_type(4)))  float     f32x4;
typedef __attribute__((ext_vector_type(4)))  unsigned  u32x4;

union AB {
  v16bf    v;
  unsigned u[8];
  u32x4    q[2];
};

// f32 pair -> packed bf16 (RNE). Use the HW cvt if this target has it.
__device__ __forceinline__ unsigned bf16pk(float lo, float hi) {
#if __has_builtin(__builtin_amdgcn_cvt_pk_bf16_f32)
  v2bf t = __builtin_amdgcn_cvt_pk_bf16_f32(lo, hi);
  return __builtin_bit_cast(unsigned, t);
#else
  unsigned a = __float_as_uint(lo), b = __float_as_uint(hi);
  a = (a + 0x7FFFu + ((a >> 16) & 1u)) >> 16;
  b = (b + 0x7FFFu + ((b >> 16) & 1u)) >> 16;
  return a | (b << 16);
#endif
}

__device__ __forceinline__ v8f wmma_bf16(const AB& a, const AB& b, v8f c) {
  return __builtin_amdgcn_wmma_f32_16x16x32_bf16(false, a.v, false, b.v,
                                                 (short)0, c, false, false);
}

// Pack 16 fp32 (A-layout chunk: [p..p+7] and [p+16..p+23]) into bf16 A operand.
// Caller passes p = rowbase + kk*32 + hi*8 so lanes 0-15 get K={0..7,16..23},
// lanes 16-31 get K={8..15,24..31} (CDNA5 16-bit A-matrix layout).
__device__ __forceinline__ void packA(AB& a, const float* p) {
  f32x4 c0 = *(const f32x4*)(p);
  f32x4 c1 = *(const f32x4*)(p + 4);
  f32x4 c2 = *(const f32x4*)(p + 16);
  f32x4 c3 = *(const f32x4*)(p + 20);
  a.u[0] = bf16pk(c0.x, c0.y); a.u[1] = bf16pk(c0.z, c0.w);
  a.u[2] = bf16pk(c1.x, c1.y); a.u[3] = bf16pk(c1.z, c1.w);
  a.u[4] = bf16pk(c2.x, c2.y); a.u[5] = bf16pk(c2.z, c2.w);
  a.u[6] = bf16pk(c3.x, c3.y); a.u[7] = bf16pk(c3.z, c3.w);
}

// xor-shuffle within the 16-lane half-groups via ds_swizzle (group-of-32 mode)
#define SWZ(x, msk) __int_as_float(__builtin_amdgcn_ds_swizzle( \
    __float_as_int(x), (((msk) << 10) | 0x1F)))

// ---------------------------------------------------------------------------
// Async DMA of one 16q x 16k x 128c fp32 z tile (128 KB) into LDS.
// Wave w moves its query rows {2w, 2w+1}; each lane carries 16 B per op,
// 32 ops per lane, tracked on ASYNCcnt (no VGPR writeback, pure streaming).
// Tile LDS layout: [qi][ki][c] linear (qi stride 8 KB, ki stride 512 B).
// ---------------------------------------------------------------------------
__device__ __forceinline__ void async_fill_z(const float* __restrict__ z,
                                             int q0, int k0n, unsigned ldsOff,
                                             int w, int lane) {
  #pragma unroll
  for (int j = 0; j < 32; ++j) {
    const int qi = 2 * w + (j >> 4);
    const unsigned tileOff = (unsigned)qi * 8192u + (unsigned)(j & 15) * 512u
                           + (unsigned)lane * 16u;
    const char* g = (const char*)z + ((size_t)(q0 + qi) << 19)   /* row: 512 KB */
                  + ((size_t)(k0n + (j & 15)) << 9)              /* key: 512 B  */
                  + (unsigned)lane * 16u;
    const unsigned l = ldsOff + tileOff;
    asm volatile("global_load_async_to_lds_b128 %0, %1, off"
                 :: "v"(l), "v"(g) : "memory");
  }
}

// ---------------------------------------------------------------------------
// Kernel 1: q = x_q@Wq+bq ; kv = x_kv@Wkv+bkv. Emits bf16 q[1024][256],
// k[1024][256], and v TRANSPOSED vt[256][1024] (so attention B-loads are
// contiguous). 64 blocks x 8 waves; 48 output col-tiles, 6 per wave.
// ---------------------------------------------------------------------------
__global__ __launch_bounds__(256) void proj_qkv(
    const float* __restrict__ xq, const float* __restrict__ xkv,
    const float* __restrict__ Wq, const float* __restrict__ bq,
    const float* __restrict__ Wkv, const float* __restrict__ bkv,
    unsigned short* __restrict__ qh, unsigned short* __restrict__ kh,
    unsigned short* __restrict__ vt)
{
  const int lane = threadIdx.x & 31;
  const int w    = threadIdx.x >> 5;
  const int m    = lane & 15;
  const int hi   = lane >> 4;
  const int r0   = blockIdx.x * 16;

  for (int tt = 0; tt < 6; ++tt) {
    const int T      = w + tt * 8;                 // 0..47
    const bool isQ   = (T < 16);
    const float* X   = isQ ? xq : xkv;
    const float* W   = isQ ? Wq : Wkv;
    const float* bia = isQ ? bq : bkv;
    const int stride = isQ ? 256 : 512;
    const int col    = (isQ ? T : (T - 16)) * 16 + m;  // lane's output column

    v8f acc = {};
    for (int kk = 0; kk < 8; ++kk) {
      AB a;
      packA(a, X + (size_t)(r0 + m) * 256 + kk * 32 + hi * 8);
      AB b;
      const float* wp = W + (size_t)(kk * 32 + hi * 8) * stride + col;
      #pragma unroll
      for (int j = 0; j < 8; ++j)
        b.u[j] = bf16pk(wp[(2 * (j & 3) + 16 * (j >> 2)) * stride], wp[(2 * (j & 3) + 16 * (j >> 2) + 1) * stride]);
      acc = wmma_bf16(a, b, acc);
    }
    const float bv = bia[col];
    #pragma unroll
    for (int r = 0; r < 8; ++r) {
      const int row = r0 + r + hi * 8;             // C-layout row
      const float o = acc[r] + bv;
      const unsigned short ob = (unsigned short)(bf16pk(o, 0.f) & 0xFFFFu);
      if (isQ)                 qh[(size_t)row * 256 + col] = ob;
      else if (col < 256)      kh[(size_t)row * 256 + col] = ob;
      else                     vt[(size_t)(col - 256) * NK + row] = ob;
    }
  }
}

// ---------------------------------------------------------------------------
// Kernel 2: fused bias-GEMM + flash attention, async-pipelined z stream.
// grid = (64 q-tiles, KSPLIT k-segments), 8 waves (one head each).
// Per k-tile iteration:
//   kick async DMA of tile it+1 into the other LDS buffer
//   s_wait_asynccnt 32  -> tile it resident; __syncthreads
//   [all waves] bias tile (16q x 16k x 8h) = z@Wb via 8 WMMAs -> LDS
//   [wave h]    s = q_h k_h^T (1 WMMA) * scale + bias + bb[h]
//               online softmax (ds_swizzle row reductions), p -> LDS
//   [wave h]    O += p @ v_h (2 WMMAs, K padded 16->32)
// Writes unnormalized partial O plus per-row (m,l) for the combine pass.
// ---------------------------------------------------------------------------
__global__ __launch_bounds__(256) void attn_bias(
    const float* __restrict__ z, const float* __restrict__ Wb,
    const float* __restrict__ bb,
    const unsigned short* __restrict__ qh, const unsigned short* __restrict__ kh,
    const unsigned short* __restrict__ vt,
    float* __restrict__ opart, float* __restrict__ mpart,
    float* __restrict__ lpart)
{
  __shared__ float zbuf[2][16 * 16 * BIASD];  // 2 x 128 KB z staging
  __shared__ float sbias[256 * 16];           // [(qi*16+ki)][head(pad16)]
  __shared__ float sp[8][256];                // per-wave p tile [qi][ki]

  const int lane = threadIdx.x & 31;
  const int w    = threadIdx.x >> 5;  // wave id == head id
  const int m    = lane & 15;
  const int hi   = lane >> 4;
  const int q0   = blockIdx.x * 16;
  const int kt0  = blockIdx.y * (64 / KSPLIT);
  const int ktn  = 64 / KSPLIT;

  // Wb (128x8, zero-padded to 128x16) B-operands for the 4 K-steps: registers.
  AB wbB[4];
  #pragma unroll
  for (int kk = 0; kk < 4; ++kk) {
    #pragma unroll
    for (int j = 0; j < 8; ++j) {
      const int c0 = kk * 32 + hi * 8 + 2 * (j & 3) + 16 * (j >> 2);
      const float f0 = (m < H) ? Wb[(size_t)c0 * H + m] : 0.f;
      const float f1 = (m < H) ? Wb[(size_t)(c0 + 1) * H + m] : 0.f;
      wbB[kk].u[j] = bf16pk(f0, f1);
    }
  }
  const float bbh = bb[w];

  // Q tile for head w, A-layout, resident in registers.
  AB qa;
  {
    const unsigned short* qp = qh + (size_t)(q0 + m) * 256 + w * 32 + hi * 8;
    qa.q[0] = *(const u32x4*)(qp);
    qa.q[1] = *(const u32x4*)(qp + 16);
  }

  float mrun[8], lrun[8];
  v8f o0 = {}, o1 = {};
  #pragma unroll
  for (int r = 0; r < 8; ++r) { mrun[r] = -3.0e38f; lrun[r] = 0.f; }

  // prologue: DMA the first z tile into buffer 0
  async_fill_z(z, q0, kt0 * 16, (unsigned)(size_t)&zbuf[0][0], w, lane);

  for (int it = 0; it < ktn; ++it) {
    const int k0 = (kt0 + it) * 16;
    const float* zt = &zbuf[it & 1][0];

    if (it + 1 < ktn) {  // overlap next tile's DMA with this tile's math
      async_fill_z(z, q0, k0 + 16, (unsigned)(size_t)&zbuf[(it + 1) & 1][0],
                   w, lane);
      asm volatile("s_wait_asynccnt 32" ::: "memory");  // prev 32 ops landed
    } else {
      asm volatile("s_wait_asynccnt 0" ::: "memory");
    }
    __syncthreads();  // tile resident for all waves; prev sbias reads done

    // ---- bias GEMM from LDS: wave w owns query rows {2w, 2w+1} ----
    #pragma unroll
    for (int t = 0; t < 2; ++t) {
      const int qi = 2 * w + t;
      const float* zr = zt + (qi * 16 + m) * BIASD;  // [qi][ki=m][*]
      v8f bacc = {};
      #pragma unroll
      for (int kk = 0; kk < 4; ++kk) {
        AB a;
        packA(a, zr + kk * 32 + hi * 8);
        bacc = wmma_bf16(a, wbB[kk], bacc);
      }
      #pragma unroll
      for (int r = 0; r < 8; ++r) {
        const int ki = r + 8 * hi;                 // C rows = keys here
        sbias[(qi * 16 + ki) * 16 + m] = bacc[r];  // col m = head
      }
    }
    __syncthreads();

    // ---- QK^T for head w: K tile in B-layout ----
    AB kb;
    {
      const unsigned short* kp = kh + (size_t)(k0 + m) * 256 + w * 32 + hi * 8;
      kb.q[0] = *(const u32x4*)(kp);
      kb.q[1] = *(const u32x4*)(kp + 16);
    }
    v8f zacc = {};
    v8f s = wmma_bf16(qa, kb, zacc);

    // ---- online softmax (rows = queries, 16-lane groups) ----
    #pragma unroll
    for (int r = 0; r < 8; ++r) {
      const int qi = r + 8 * hi;                 // C-layout row
      const int ki = m;                          // C-layout col
      float sv = s[r] * QSCALE + sbias[(qi * 16 + ki) * 16 + w] + bbh;
      float mx = sv;
      mx = fmaxf(mx, SWZ(mx, 1)); mx = fmaxf(mx, SWZ(mx, 2));
      mx = fmaxf(mx, SWZ(mx, 4)); mx = fmaxf(mx, SWZ(mx, 8));
      const float mnew = fmaxf(mrun[r], mx);
      const float pv = __expf(sv - mnew);
      float sum = pv;
      sum += SWZ(sum, 1); sum += SWZ(sum, 2);
      sum += SWZ(sum, 4); sum += SWZ(sum, 8);
      const float alpha = __expf(mrun[r] - mnew);
      lrun[r] = lrun[r] * alpha + sum;
      o0[r] *= alpha;
      o1[r] *= alpha;
      mrun[r] = mnew;
      sp[w][qi * 16 + ki] = pv;   // wave-private stage; DS is in-order per wave
    }

    // ---- O += p @ v : A = p (K 16 real + 16 zero), B = vt rows ----
    AB pa;
    {
      const float* pp = &sp[w][m * 16 + hi * 8];
      f32x4 c0 = *(const f32x4*)(pp);
      f32x4 c1 = *(const f32x4*)(pp + 4);
      pa.u[0] = bf16pk(c0.x, c0.y); pa.u[1] = bf16pk(c0.z, c0.w);
      pa.u[2] = bf16pk(c1.x, c1.y); pa.u[3] = bf16pk(c1.z, c1.w);
      pa.u[4] = 0; pa.u[5] = 0; pa.u[6] = 0; pa.u[7] = 0;  // K=16..31 pad
    }
    AB vb0, vb1;
    {
      u32x4 zz = {};
      const unsigned short* vp0 = vt + (size_t)(w * 32 + m) * NK + k0;
      const unsigned short* vp1 = vt + (size_t)(w * 32 + 16 + m) * NK + k0;
      {                                           // K=8*hi..8*hi+7 real, K>=16 pad
        vb0.q[0] = *(const u32x4*)(vp0 + hi * 8); vb0.q[1] = zz;
        vb1.q[0] = *(const u32x4*)(vp1 + hi * 8); vb1.q[1] = zz;
      }
    }
    o0 = wmma_bf16(pa, vb0, o0);
    o1 = wmma_bf16(pa, vb1, o1);
  }

  // ---- epilogue: unnormalized partial O + (m,l) stats ----
  const size_t obase = ((size_t)blockIdx.y * NQ + q0) * (H * Dh);
  #pragma unroll
  for (int r = 0; r < 8; ++r) {
    const int qrow = r + 8 * hi;
    opart[obase + (size_t)qrow * 256 + w * 32 + m]      = o0[r];
    opart[obase + (size_t)qrow * 256 + w * 32 + 16 + m] = o1[r];
    if (m == 0) {
      mpart[((size_t)blockIdx.y * H + w) * NQ + q0 + qrow] = mrun[r];
      lpart[((size_t)blockIdx.y * H + w) * NQ + q0 + qrow] = lrun[r];
    }
  }
}

// ---------------------------------------------------------------------------
// Kernel 3: combine the KSPLIT flash partials into normalized O.
// ---------------------------------------------------------------------------
__global__ __launch_bounds__(256) void combine_kern(
    const float* __restrict__ opart, const float* __restrict__ mpart,
    const float* __restrict__ lpart, float* __restrict__ ofull)
{
  const int idx = blockIdx.x * blockDim.x + threadIdx.x;
  if (idx >= NQ * 256) return;
  const int q = idx >> 8, c = idx & 255, h = c >> 5;
  float mv[KSPLIT];
  float mstar = -3.0e38f;
  #pragma unroll
  for (int s = 0; s < KSPLIT; ++s) {
    mv[s] = mpart[((size_t)s * H + h) * NQ + q];
    mstar = fmaxf(mstar, mv[s]);
  }
  float denom = 0.f, acc = 0.f;
  #pragma unroll
  for (int s = 0; s < KSPLIT; ++s) {
    const float e = __expf(mv[s] - mstar);
    denom += lpart[((size_t)s * H + h) * NQ + q] * e;
    acc   += opart[((size_t)s * NQ + q) * 256 + c] * e;
  }
  ofull[idx] = acc / denom;
}

// ---------------------------------------------------------------------------
// Kernel 4: out = O @ Wp + bp  (1024x256x256). 64 blocks x 8 waves, 2
// col-tiles per wave.
// ---------------------------------------------------------------------------
__global__ __launch_bounds__(256) void proj_out(
    const float* __restrict__ ofull, const float* __restrict__ Wp,
    const float* __restrict__ bp, float* __restrict__ out)
{
  const int lane = threadIdx.x & 31;
  const int w    = threadIdx.x >> 5;
  const int m    = lane & 15;
  const int hi   = lane >> 4;
  const int r0   = blockIdx.x * 16;

  for (int tt = 0; tt < 2; ++tt) {
    const int T   = w + tt * 8;
    const int col = T * 16 + m;
    v8f acc = {};
    for (int kk = 0; kk < 8; ++kk) {
      AB a;
      packA(a, ofull + (size_t)(r0 + m) * 256 + kk * 32 + hi * 8);
      AB b;
      const float* wp = Wp + (size_t)(kk * 32 + hi * 8) * 256 + col;
      #pragma unroll
      for (int j = 0; j < 8; ++j)
        b.u[j] = bf16pk(wp[(2 * (j & 3) + 16 * (j >> 2)) * 256], wp[(2 * (j & 3) + 16 * (j >> 2) + 1) * 256]);
      acc = wmma_bf16(a, b, acc);
    }
    const float bv = bp[col];
    #pragma unroll
    for (int r = 0; r < 8; ++r)
      out[(size_t)(r0 + r + hi * 8) * 256 + col] = acc[r] + bv;
  }
}

// ---------------------------------------------------------------------------
extern "C" void kernel_launch(void* const* d_in, const int* in_sizes, int n_in,
                              void* d_out, int out_size, void* d_ws,
                              size_t ws_size, hipStream_t stream) {
  const float* x_q  = (const float*)d_in[0];
  const float* x_kv = (const float*)d_in[1];
  const float* z    = (const float*)d_in[2];
  const float* Wq   = (const float*)d_in[3];
  const float* bq   = (const float*)d_in[4];
  const float* Wkv  = (const float*)d_in[5];
  const float* bkv  = (const float*)d_in[6];
  const float* Wb   = (const float*)d_in[7];
  const float* bb   = (const float*)d_in[8];
  const float* Wp   = (const float*)d_in[9];
  const float* bp   = (const float*)d_in[10];
  float* out = (float*)d_out;

  char* ws = (char*)d_ws;
  size_t off = 0;
  unsigned short* qh = (unsigned short*)(ws + off); off += (size_t)NQ * 256 * 2;
  unsigned short* kh = (unsigned short*)(ws + off); off += (size_t)NK * 256 * 2;
  unsigned short* vt = (unsigned short*)(ws + off); off += (size_t)256 * NK * 2;
  float* opart = (float*)(ws + off); off += (size_t)KSPLIT * NQ * 256 * 4;
  float* mpart = (float*)(ws + off); off += (size_t)KSPLIT * H * NQ * 4;
  float* lpart = (float*)(ws + off); off += (size_t)KSPLIT * H * NQ * 4;
  float* ofull = (float*)(ws + off); off += (size_t)NQ * 256 * 4;

  proj_qkv<<<64, 256, 0, stream>>>(x_q, x_kv, Wq, bq, Wkv, bkv, qh, kh, vt);
  attn_bias<<<dim3(64, KSPLIT), 256, 0, stream>>>(z, Wb, bb, qh, kh, vt,
                                                  opart, mpart, lpart);
  combine_kern<<<(NQ * 256 + 255) / 256, 256, 0, stream>>>(opart, mpart,
                                                           lpart, ofull);
  proj_out<<<64, 256, 0, stream>>>(ofull, Wp, bp, out);
}
